// GPTLanguageModel_67808943669592
// MI455X (gfx1250) — compile-verified, matrix operand order checked by reading, not run on hardware
//
#include <hip/hip_runtime.h>
#include <hip/hip_bf16.h>

// ---------------------------------------------------------------------------
// GPT forward (L=4, H=16, D=1024, HS=64, V=32000, T=1024, B=2) on gfx1250.
// f16 operands everywhere (converted once, outside hot loops), fp32 accum.
// GEMM: double-buffered LDS tiles staged with GLOBAL_LOAD_ASYNC_TO_LDS_B128,
// compute with v_wmma_f32_16x16x32_f16; last K-step peeled so the steady
// loop is branch-free.
// ---------------------------------------------------------------------------

#define L_  4
#define H_  16
#define D_  1024
#define HS_ 64
#define V_  32000
#define T_  1024
#define B_  2
#define ROWS_ (B_ * T_)          // 2048

typedef __attribute__((ext_vector_type(16))) _Float16 v16h;
typedef __attribute__((ext_vector_type(8)))  _Float16 v8h;
typedef __attribute__((ext_vector_type(8)))  float    v8f;
typedef int v4i_ __attribute__((vector_size(16)));
typedef __attribute__((address_space(1))) v4i_ gv4i;
typedef __attribute__((address_space(3))) v4i_ lv4i;

union Frag16 { v16h v; v8h h8[2]; };

__device__ __forceinline__ v16h frag_ld(const _Float16* p0, const _Float16* p1) {
    Frag16 f;
    f.h8[0] = *reinterpret_cast<const v8h*>(p0);
    f.h8[1] = *reinterpret_cast<const v8h*>(p1);
    return f.v;
}

#define WMMA_F16(A, B, C) \
    __builtin_amdgcn_wmma_f32_16x16x32_f16(false, (A), false, (B), (short)0, (C), false, false)

// 16-byte async copy global -> LDS (tracked by ASYNCcnt)
__device__ __forceinline__ void async_copy16(const void* g, void* l) {
    __builtin_amdgcn_global_load_async_to_lds_b128((gv4i*)g, (lv4i*)l, 0, 0);
}
#define WAIT_ASYNC_LE4() asm volatile("s_wait_asynccnt 0x4" ::: "memory")
#define WAIT_ASYNC_0()   asm volatile("s_wait_asynccnt 0x0" ::: "memory")

// ---------------------------------------------------------------------------
// 0) Weight transpose + f16 convert:  WT[n][k] = (f16) W[k][n]
//    headblk: W stored as [n/64][k][n%64] (the [H][D][HS] qkv layout)
// ---------------------------------------------------------------------------
__global__ __launch_bounds__(256) void wt_kernel(const float* __restrict__ W,
                                                 _Float16* __restrict__ WT,
                                                 int K, int N, int headblk) {
    __shared__ _Float16 tile[32][40];
    const int tid = threadIdx.x;
    const int n0 = blockIdx.x * 32;
    const int k0 = blockIdx.y * 32;
    const int tx = tid & 31;     // n_local
    const int ty = tid >> 5;     // 0..7
#pragma unroll
    for (int i = 0; i < 4; ++i) {
        const int kl = ty + i * 8;
        const int n = n0 + tx, kg = k0 + kl;
        const size_t idx = headblk ? ((size_t)(n >> 6) * K + kg) * 64 + (n & 63)
                                   : (size_t)kg * N + n;
        tile[tx][kl] = (_Float16)W[idx];
    }
    __syncthreads();
    const int r  = tid >> 3;         // n_local
    const int c4 = (tid & 7) * 4;    // k_local
    _Float16* dst = &WT[(size_t)(n0 + r) * K + k0 + c4];
    dst[0] = tile[r][c4 + 0];
    dst[1] = tile[r][c4 + 1];
    dst[2] = tile[r][c4 + 2];
    dst[3] = tile[r][c4 + 3];
}

// ---------------------------------------------------------------------------
// 1) Embedding: x[b,t,:] = tok_emb[idx[b,t],:] + pos_emb[t,:]   (fp32 out)
// ---------------------------------------------------------------------------
__global__ __launch_bounds__(256) void embed_kernel(const int* __restrict__ ctx,
                                                    const float* __restrict__ tok,
                                                    const float* __restrict__ pos,
                                                    float* __restrict__ x) {
    const int row = blockIdx.x;
    const int t   = row & (T_ - 1);
    const int id  = ctx[row];
    const size_t xo = (size_t)row * D_;
    const size_t to = (size_t)id * D_;
    const size_t po = (size_t)t * D_;
    for (int j = threadIdx.x; j < D_; j += 256)
        x[xo + j] = tok[to + j] + pos[po + j];
}

// ---------------------------------------------------------------------------
// 2) LayerNorm: fp32 in, f16 out (feeds GEMMs only)
// ---------------------------------------------------------------------------
__global__ __launch_bounds__(256) void ln_kernel(const float* __restrict__ in,
                                                 const float* __restrict__ g,
                                                 const float* __restrict__ bta,
                                                 _Float16* __restrict__ out) {
    __shared__ float s1[256];
    __shared__ float s2[256];
    const int row = blockIdx.x;
    const int tid = threadIdx.x;
    const size_t ro = (size_t)row * D_;
    float a = 0.f, b = 0.f;
    for (int j = tid; j < D_; j += 256) {
        float v = in[ro + j];
        a += v; b += v * v;
    }
    s1[tid] = a; s2[tid] = b;
    __syncthreads();
    for (int s = 128; s > 0; s >>= 1) {
        if (tid < s) { s1[tid] += s1[tid + s]; s2[tid] += s2[tid + s]; }
        __syncthreads();
    }
    const float mean = s1[0] * (1.0f / D_);
    const float var  = s2[0] * (1.0f / D_) - mean * mean;
    const float rstd = rsqrtf(var + 1e-5f);
    for (int j = tid; j < D_; j += 256)
        out[ro + j] = (_Float16)((in[ro + j] - mean) * rstd * g[j] + bta[j]);
}

// ---------------------------------------------------------------------------
// 3) WMMA GEMM: C[M,N] = A[M,K](f16) * WT[N,K](f16, pre-transposed)
//    + bias + residual(f32), optional ReLU; C is f32 or f16 (flag).
//    256 threads = 8 waves (2x4); BM=BN=128, BK=32; double-buffered LDS
//    staged entirely with async b128 copies; last K-step peeled.
// ---------------------------------------------------------------------------
#define GF_RELU   1
#define GF_OUTF16 2
#define BM 128
#define BN 128
#define BK 32
#define APITCH 40   // halves; 80B row pitch keeps 16B alignment

__global__ __launch_bounds__(256) void gemm_f16_kernel(const _Float16* __restrict__ A,
                                                       const _Float16* __restrict__ WT,
                                                       const float* __restrict__ bias,
                                                       const float* __restrict__ res,
                                                       void* __restrict__ Cout,
                                                       int M, int N, int K, int flags) {
    __shared__ __align__(16) _Float16 As[2][BM][APITCH];
    __shared__ __align__(16) _Float16 Bs[2][BN][APITCH];

    const int tid  = threadIdx.x;
    const int wid  = tid >> 5;
    const int lane = tid & 31;
    const int m0 = blockIdx.y * BM;
    const int n0 = blockIdx.x * BN;
    const int wm = (wid >> 2) * 64;
    const int wn = (wid & 3) * 32;
    const int lrow = lane & 15;
    const int kb8  = (lane >> 4) * 8;

    // per-thread staging descriptors (two 16B chunks per tile per matrix)
    const int r0  = tid >> 2;
    const int k80 = (tid & 3) * 8;
    const int r1  = (tid + 256) >> 2;
    const int k81 = ((tid + 256) & 3) * 8;
    const _Float16* ga0 = A  + (size_t)(m0 + r0) * K + k80;
    const _Float16* ga1 = A  + (size_t)(m0 + r1) * K + k81;
    const _Float16* gb0 = WT + (size_t)(n0 + r0) * K + k80;
    const _Float16* gb1 = WT + (size_t)(n0 + r1) * K + k81;

    auto stage = [&](int buf, int kk) {
        async_copy16(ga0 + kk, &As[buf][r0][k80]);
        async_copy16(gb0 + kk, &Bs[buf][r0][k80]);
        async_copy16(ga1 + kk, &As[buf][r1][k81]);
        async_copy16(gb1 + kk, &Bs[buf][r1][k81]);
    };

    v8f acc[4][2];
    const v8f vzero = {0.f, 0.f, 0.f, 0.f, 0.f, 0.f, 0.f, 0.f};
#pragma unroll
    for (int mt = 0; mt < 4; ++mt)
#pragma unroll
        for (int nt = 0; nt < 2; ++nt) acc[mt][nt] = vzero;

    auto compute = [&](int buf) {
        v16h af[4], bf[2];
#pragma unroll
        for (int mt = 0; mt < 4; ++mt) {
            const int r = wm + mt * 16 + lrow;
            af[mt] = frag_ld(&As[buf][r][kb8], &As[buf][r][16 + kb8]);
        }
#pragma unroll
        for (int nt = 0; nt < 2; ++nt) {
            const int n = wn + nt * 16 + lrow;
            bf[nt] = frag_ld(&Bs[buf][n][kb8], &Bs[buf][n][16 + kb8]);
        }
#pragma unroll
        for (int mt = 0; mt < 4; ++mt)
#pragma unroll
            for (int nt = 0; nt < 2; ++nt)
                acc[mt][nt] = WMMA_F16(af[mt], bf[nt], acc[mt][nt]);
    };

    const int nk = K / BK;
    stage(0, 0);
    // steady state: branch-free pipeline
    for (int it = 0; it < nk - 1; ++it) {
        const int buf = it & 1;
        stage(buf ^ 1, (it + 1) * BK);
        WAIT_ASYNC_LE4();            // tile `it` landed; next tile in flight
        __syncthreads();
        compute(buf);
        __syncthreads();
    }
    // peeled last K-step
    WAIT_ASYNC_0();
    __syncthreads();
    compute((nk - 1) & 1);

    // epilogue: C lane mapping M = r + 8*(lane>=16), N = lane&15
    const int ml8 = (lane >> 4) * 8;
    float* Cf = (float*)Cout;
    _Float16* Ch = (_Float16*)Cout;
#pragma unroll
    for (int mt = 0; mt < 4; ++mt) {
#pragma unroll
        for (int nt = 0; nt < 2; ++nt) {
            const int col = n0 + wn + nt * 16 + lrow;
            const float bv = bias ? bias[col] : 0.f;
#pragma unroll
            for (int r = 0; r < 8; ++r) {
                const int row = m0 + wm + mt * 16 + ml8 + r;
                float val = acc[mt][nt][r] + bv;
                if (res) val += res[(size_t)row * N + col];
                if (flags & GF_RELU) val = fmaxf(val, 0.f);
                if (flags & GF_OUTF16) Ch[(size_t)row * N + col] = (_Float16)val;
                else                   Cf[(size_t)row * N + col] = val;
            }
        }
    }
}

// ---------------------------------------------------------------------------
// 4) Flash attention: one wave per (b, h, 16-query block); f16 q/k/v in
//    [B*T, H*HS]; K tiles async-staged; S and O via WMMA; f16 out.
// ---------------------------------------------------------------------------
__global__ __launch_bounds__(32) void attn_kernel(const _Float16* __restrict__ q,
                                                  const _Float16* __restrict__ k,
                                                  const _Float16* __restrict__ v,
                                                  _Float16* __restrict__ o,
                                                  float scale) {
    __shared__ __align__(16) _Float16 Qs[16][72];   // [row][e]
    __shared__ __align__(16) _Float16 Ks[32][72];   // [key][e]
    __shared__ __align__(16) _Float16 Vt[64][40];   // [e][key]
    __shared__ __align__(16) _Float16 Ps[16][40];   // [row][key]

    const int lane = threadIdx.x;
    const int qb   = blockIdx.x;           // 0..63
    const int bh   = blockIdx.y;           // 0..31
    const int bb   = bh >> 4;
    const int hh   = bh & 15;
    const int q0   = qb * 16;
    const size_t base = (size_t)bb * T_ * D_ + (size_t)hh * HS_;

    for (int c = lane; c < 128; c += 32) {          // 16 rows x 8 chunks
        const int r = c >> 3, e8 = (c & 7) * 8;
        *reinterpret_cast<v8h*>(&Qs[r][e8]) =
            *reinterpret_cast<const v8h*>(&q[base + (size_t)(q0 + r) * D_ + e8]);
    }
    __syncthreads();

    const int lrow = lane & 15;
    const int kb8  = (lane >> 4) * 8;
    const int ml8  = (lane >> 4) * 8;

    v16h aq[2];
#pragma unroll
    for (int s = 0; s < 2; ++s)
        aq[s] = frag_ld(&Qs[lrow][32 * s + kb8], &Qs[lrow][32 * s + 16 + kb8]);

    const v8f vzero = {0.f, 0.f, 0.f, 0.f, 0.f, 0.f, 0.f, 0.f};
    v8f oacc[4];
#pragma unroll
    for (int nt = 0; nt < 4; ++nt) oacc[nt] = vzero;
    float rmax[8], rsum[8];
#pragma unroll
    for (int r = 0; r < 8; ++r) { rmax[r] = -1e30f; rsum[r] = 0.f; }

    const int nkb = (qb >> 1) + 1;
    for (int kbI = 0; kbI < nkb; ++kbI) {
        const int kt0 = kbI * 32;
        // K tile: pure async copies (32 keys x 8 chunks)
        for (int c = lane; c < 256; c += 32) {
            const int kr = c >> 3, e8 = (c & 7) * 8;
            async_copy16(&k[base + (size_t)(kt0 + kr) * D_ + e8], &Ks[kr][e8]);
        }
        // V tile: transposed scatter [e][key]
        for (int c = lane; c < 256; c += 32) {
            const int kr = c >> 3, e8 = (c & 7) * 8;
            const v8h vv = *reinterpret_cast<const v8h*>(&v[base + (size_t)(kt0 + kr) * D_ + e8]);
#pragma unroll
            for (int j = 0; j < 8; ++j) Vt[e8 + j][kr] = vv[j];
        }
        WAIT_ASYNC_0();
        __syncthreads();

        // S = Q * K^T
        v8f s0 = vzero, s1 = vzero;
#pragma unroll
        for (int s = 0; s < 2; ++s) {
            v16h bk0 = frag_ld(&Ks[lrow][32 * s + kb8],      &Ks[lrow][32 * s + 16 + kb8]);
            v16h bk1 = frag_ld(&Ks[16 + lrow][32 * s + kb8], &Ks[16 + lrow][32 * s + 16 + kb8]);
            s0 = WMMA_F16(aq[s], bk0, s0);
            s1 = WMMA_F16(aq[s], bk1, s1);
        }

        // online softmax
#pragma unroll
        for (int r = 0; r < 8; ++r) {
            const int mq = q0 + ml8 + r;
            float v0 = s0[r] * scale;
            float v1 = s1[r] * scale;
            if (kt0 + lrow > mq)      v0 = -1e30f;
            if (kt0 + 16 + lrow > mq) v1 = -1e30f;
            float t = fmaxf(v0, v1);
#pragma unroll
            for (int m = 1; m < 16; m <<= 1) t = fmaxf(t, __shfl_xor(t, m, 32));
            const float nm = fmaxf(rmax[r], t);
            const float p0 = __expf(v0 - nm);
            const float p1 = __expf(v1 - nm);
            float ssum = p0 + p1;
#pragma unroll
            for (int m = 1; m < 16; m <<= 1) ssum += __shfl_xor(ssum, m, 32);
            const float corr = __expf(rmax[r] - nm);
            rsum[r] = rsum[r] * corr + ssum;
            rmax[r] = nm;
#pragma unroll
            for (int nt = 0; nt < 4; ++nt) oacc[nt][r] *= corr;
            Ps[ml8 + r][lrow]      = (_Float16)p0;
            Ps[ml8 + r][16 + lrow] = (_Float16)p1;
        }
        __syncthreads();

        // O += P * V
        v16h ap = frag_ld(&Ps[lrow][kb8], &Ps[lrow][16 + kb8]);
#pragma unroll
        for (int nt = 0; nt < 4; ++nt) {
            const int ne = nt * 16 + lrow;
            v16h bv = frag_ld(&Vt[ne][kb8], &Vt[ne][16 + kb8]);
            oacc[nt] = WMMA_F16(ap, bv, oacc[nt]);
        }
        __syncthreads();
    }

#pragma unroll
    for (int nt = 0; nt < 4; ++nt)
#pragma unroll
        for (int r = 0; r < 8; ++r) {
            const int row = q0 + ml8 + r;
            const int col = nt * 16 + lrow;
            o[base + (size_t)row * D_ + col] = (_Float16)(oacc[nt][r] / rsum[r]);
        }
}

// ---------------------------------------------------------------------------
// 5) Per-row NLL + deterministic reduction
// ---------------------------------------------------------------------------
__global__ __launch_bounds__(256) void nll_kernel(const float* __restrict__ logits,
                                                  const int* __restrict__ tgt,
                                                  float* __restrict__ nll) {
    __shared__ float red[256];
    const int row = blockIdx.x;
    const int tid = threadIdx.x;
    const float* lr = logits + (size_t)row * V_;
    float mx = -1e30f;
    for (int j = tid; j < V_; j += 256) mx = fmaxf(mx, lr[j]);
    red[tid] = mx;
    __syncthreads();
    for (int s = 128; s > 0; s >>= 1) {
        if (tid < s) red[tid] = fmaxf(red[tid], red[tid + s]);
        __syncthreads();
    }
    mx = red[0];
    __syncthreads();
    float sm = 0.f;
    for (int j = tid; j < V_; j += 256) sm += __expf(lr[j] - mx);
    red[tid] = sm;
    __syncthreads();
    for (int s = 128; s > 0; s >>= 1) {
        if (tid < s) red[tid] += red[tid + s];
        __syncthreads();
    }
    if (tid == 0)
        nll[row] = mx + __logf(red[0]) - lr[tgt[row]];
}

__global__ __launch_bounds__(256) void loss_finalize_kernel(const float* __restrict__ nll,
                                                            float* __restrict__ out_loss) {
    __shared__ float red[256];
    const int tid = threadIdx.x;
    float s = 0.f;
    for (int j = tid; j < ROWS_; j += 256) s += nll[j];
    red[tid] = s;
    __syncthreads();
    for (int st = 128; st > 0; st >>= 1) {
        if (tid < st) red[tid] += red[tid + st];
        __syncthreads();
    }
    if (tid == 0) out_loss[0] = red[0] / (float)ROWS_;
}

// ---------------------------------------------------------------------------
// Host orchestration
// ---------------------------------------------------------------------------
extern "C" void kernel_launch(void* const* d_in, const int* in_sizes, int n_in,
                              void* d_out, int out_size, void* d_ws, size_t ws_size,
                              hipStream_t stream) {
    (void)in_sizes; (void)n_in; (void)out_size; (void)ws_size;

    const int*   ctx   = (const int*)  d_in[0];
    const int*   tgt   = (const int*)  d_in[1];
    const float* tok   = (const float*)d_in[2];
    const float* pos   = (const float*)d_in[3];
    const float* wq    = (const float*)d_in[4];
    const float* wk    = (const float*)d_in[5];
    const float* wv    = (const float*)d_in[6];
    const float* wo    = (const float*)d_in[7];
    const float* bo    = (const float*)d_in[8];
    const float* ln1g  = (const float*)d_in[9];
    const float* ln1b  = (const float*)d_in[10];
    const float* w1    = (const float*)d_in[11];
    const float* b1    = (const float*)d_in[12];
    const float* w2    = (const float*)d_in[13];
    const float* b2    = (const float*)d_in[14];
    const float* ln2g  = (const float*)d_in[15];
    const float* ln2b  = (const float*)d_in[16];
    const float* lnfg  = (const float*)d_in[17];
    const float* lnfb  = (const float*)d_in[18];
    const float* lmw   = (const float*)d_in[19];
    const float* lmb   = (const float*)d_in[20];

    // ---- workspace carve ----
    char* p = (char*)d_ws;
    auto alloc = [&](size_t bytes) -> void* {
        void* r = (void*)p;
        p += (bytes + 255) & ~(size_t)255;
        return r;
    };
    const size_t RD = (size_t)ROWS_ * D_;
    float*     x    = (float*)    alloc(RD * 4);
    _Float16*  h    = (_Float16*) alloc(RD * 2);
    _Float16*  qb   = (_Float16*) alloc(RD * 2);
    _Float16*  kb   = (_Float16*) alloc(RD * 2);
    _Float16*  vb   = (_Float16*) alloc(RD * 2);
    _Float16*  ob   = (_Float16*) alloc(RD * 2);
    _Float16*  mid  = (_Float16*) alloc((size_t)ROWS_ * 4 * D_ * 2);
    float*     nll  = (float*)    alloc(ROWS_ * 4);
    _Float16*  wqT  = (_Float16*) alloc((size_t)L_ * D_ * D_ * 2);
    _Float16*  wkT  = (_Float16*) alloc((size_t)L_ * D_ * D_ * 2);
    _Float16*  wvT  = (_Float16*) alloc((size_t)L_ * D_ * D_ * 2);
    _Float16*  woT  = (_Float16*) alloc((size_t)L_ * D_ * D_ * 2);
    _Float16*  w1T  = (_Float16*) alloc((size_t)L_ * D_ * 4 * D_ * 2);
    _Float16*  w2T  = (_Float16*) alloc((size_t)L_ * 4 * D_ * D_ * 2);
    _Float16*  lmT  = (_Float16*) alloc((size_t)D_ * V_ * 2);

    float* logits = (float*)d_out;
    float* loss   = logits + (size_t)ROWS_ * V_;

    const dim3 blk256(256), blk32(32);

    // ---- weight f16 transposes (once per launch) ----
    for (int l = 0; l < L_; ++l) {
        const size_t oW  = (size_t)l * D_ * D_;
        const size_t oM  = (size_t)l * D_ * 4 * D_;
        wt_kernel<<<dim3(D_/32, D_/32), blk256, 0, stream>>>(wq + oW, wqT + oW, D_, D_, 1);
        wt_kernel<<<dim3(D_/32, D_/32), blk256, 0, stream>>>(wk + oW, wkT + oW, D_, D_, 1);
        wt_kernel<<<dim3(D_/32, D_/32), blk256, 0, stream>>>(wv + oW, wvT + oW, D_, D_, 1);
        wt_kernel<<<dim3(D_/32, D_/32), blk256, 0, stream>>>(wo + oW, woT + oW, D_, D_, 0);
        wt_kernel<<<dim3(4*D_/32, D_/32), blk256, 0, stream>>>(w1 + oM, w1T + oM, D_, 4*D_, 0);
        wt_kernel<<<dim3(D_/32, 4*D_/32), blk256, 0, stream>>>(w2 + oM, w2T + oM, 4*D_, D_, 0);
    }
    wt_kernel<<<dim3(V_/32, D_/32), blk256, 0, stream>>>(lmw, lmT, D_, V_, 0);

    const dim3 gD (D_  / BN, ROWS_ / BM);
    const dim3 g4D(4*D_ / BN, ROWS_ / BM);
    const dim3 gV (V_  / BN, ROWS_ / BM);
    const float scale = 0.03125f;   // D^-0.5 = 1/32 (ref scales by emb dim)

    embed_kernel<<<ROWS_, blk256, 0, stream>>>(ctx, tok, pos, x);

    for (int l = 0; l < L_; ++l) {
        const size_t oW  = (size_t)l * D_ * D_;
        const size_t oM  = (size_t)l * D_ * 4 * D_;
        const size_t oD  = (size_t)l * D_;
        const size_t o4D = (size_t)l * 4 * D_;

        ln_kernel<<<ROWS_, blk256, 0, stream>>>(x, ln1g + oD, ln1b + oD, h);

        gemm_f16_kernel<<<gD, blk256, 0, stream>>>(h, wqT + oW, nullptr, nullptr, qb,
                                                   ROWS_, D_, D_, GF_OUTF16);
        gemm_f16_kernel<<<gD, blk256, 0, stream>>>(h, wkT + oW, nullptr, nullptr, kb,
                                                   ROWS_, D_, D_, GF_OUTF16);
        gemm_f16_kernel<<<gD, blk256, 0, stream>>>(h, wvT + oW, nullptr, nullptr, vb,
                                                   ROWS_, D_, D_, GF_OUTF16);

        attn_kernel<<<dim3(T_ / 16, B_ * H_), blk32, 0, stream>>>(qb, kb, vb, ob, scale);

        gemm_f16_kernel<<<gD, blk256, 0, stream>>>(ob, woT + oW, bo + oD, x, x,
                                                   ROWS_, D_, D_, 0);

        ln_kernel<<<ROWS_, blk256, 0, stream>>>(x, ln2g + oD, ln2b + oD, h);

        gemm_f16_kernel<<<g4D, blk256, 0, stream>>>(h, w1T + oM, b1 + o4D, nullptr, mid,
                                                    ROWS_, 4 * D_, D_, GF_RELU | GF_OUTF16);
        gemm_f16_kernel<<<gD, blk256, 0, stream>>>(mid, w2T + oM, b2 + oD, x, x,
                                                   ROWS_, D_, 4 * D_, 0);
    }

    ln_kernel<<<ROWS_, blk256, 0, stream>>>(x, lnfg, lnfb, h);

    gemm_f16_kernel<<<gV, blk256, 0, stream>>>(h, lmT, lmb, nullptr, logits,
                                               ROWS_, V_, D_, 0);

    nll_kernel<<<ROWS_, blk256, 0, stream>>>(logits, tgt, nll);
    loss_finalize_kernel<<<1, blk256, 0, stream>>>(nll, loss);
}
